// S4_37993280700537
// MI455X (gfx1250) — matrix operand order repaired and checked by reading, not compiled, not run on hardware
//
#include <hip/hip_runtime.h>
#include <hip/hip_bf16.h>
#include <math.h>

// S4 forward for MI455X (gfx1250, wave32, WMMA, async-LDS).
// B=8, H=512, N=64, L=2048, CH=1, chunk Q=64, chunks NC=32.
//
// Pipeline:
//  K1 s4_precompute : per (h,n) constants: w=exp(dt*A), C_disc, k[0..63],
//                     Wr[n][j]=w^{63-j}, Cw[r][n]=2*Cd*w^{r+1} (Im pre-negated), w^64.
//  K2 s4_chunk_scan : per (b,h) wave: Y = T@U + CwRe@XpRe + CwImNeg@XpIm (6 WMMA/tile),
//                     S=Wr@U (WMMA) + 32-step complex chunk scan in between.
//                     u staged to LDS with global_load_async_to_lds_b128 (ASYNCcnt).
//                     Epilogue: +D*u, exact GELU, store g as f16.
//  K3 s4_outproj    : z = W_out @ g (f16 WMMA, f32 acc) + b, fused GLU h1*sigmoid(h2),
//                     with global_prefetch of the next g panel.

typedef __attribute__((ext_vector_type(16))) _Float16 v16h;
typedef __attribute__((ext_vector_type(8)))  float    v8f;
typedef __attribute__((ext_vector_type(8)))  _Float16 v8h;

#define S4_B   8
#define S4_H   512
#define S4_N   64
#define S4_L   2048
#define S4_Q   64
#define S4_NC  32

static __device__ __forceinline__ v8f wmma_f16(v16h a, v16h b, v8f c) {
  // D = A(16x32 f16) * B(32x16 f16) + C(16x16 f32)
  return __builtin_amdgcn_wmma_f32_16x16x32_f16(false, a, false, b, (short)0, c,
                                                false, false);
}

// A fragment (16x32 f16) from row-major f16 source; m = lane&15, kh = lane>>4.
static __device__ __forceinline__ v16h load_a_f16(const _Float16* __restrict__ src,
                                                  int m, int k0, int kh, int stride) {
  v16h a;
#pragma unroll
  for (int v = 0; v < 8; ++v) {
    int k = k0 + ((v >= 4) ? 16 : 0) + kh * 8 + (v & 3) * 2;
    a[2 * v]     = src[(size_t)m * stride + k];
    a[2 * v + 1] = src[(size_t)m * stride + k + 1];
  }
  return a;
}

// A fragment from row-major f32 source, converted to f16.
static __device__ __forceinline__ v16h load_a_f32(const float* __restrict__ src,
                                                  int m, int k0, int kh, int stride) {
  v16h a;
#pragma unroll
  for (int v = 0; v < 8; ++v) {
    int k = k0 + ((v >= 4) ? 16 : 0) + kh * 8 + (v & 3) * 2;
    a[2 * v]     = (_Float16)src[(size_t)m * stride + k];
    a[2 * v + 1] = (_Float16)src[(size_t)m * stride + k + 1];
  }
  return a;
}

// B fragment (32x16 f16) from column-major f16 storage: (k,col) at col*stride + k.
static __device__ __forceinline__ v16h load_b_cm(const _Float16* __restrict__ src,
                                                 int col, int k0, int kh, int stride) {
  v16h b;
#pragma unroll
  for (int v = 0; v < 8; ++v) {
    int k = k0 + ((v >= 4) ? 16 : 0) + kh * 8 + (v & 3) * 2;
    b[2 * v]     = src[col * stride + k];
    b[2 * v + 1] = src[col * stride + k + 1];
  }
  return b;
}

// B fragment (32x16) from column-major f32 storage, converted to f16.
static __device__ __forceinline__ v16h load_b_cm_f32(const float* __restrict__ src,
                                                     int col, int k0, int kh, int stride) {
  v16h b;
#pragma unroll
  for (int v = 0; v < 8; ++v) {
    int k = k0 + ((v >= 4) ? 16 : 0) + kh * 8 + (v & 3) * 2;
    b[2 * v]     = (_Float16)src[col * stride + k];
    b[2 * v + 1] = (_Float16)src[col * stride + k + 1];
  }
  return b;
}

// ---------------------------------------------------------------------------
// Kernel 1: per-h precompute.  grid = H, block = N (=64).
// ---------------------------------------------------------------------------
__global__ __launch_bounds__(64) void s4_precompute(
    const float* __restrict__ log_dt,
    const float* __restrict__ A_re, const float* __restrict__ A_im,
    const float* __restrict__ C_re, const float* __restrict__ C_im,
    float* __restrict__ kq, float* __restrict__ wQre, float* __restrict__ wQim,
    _Float16* __restrict__ WrRe, _Float16* __restrict__ WrIm,
    _Float16* __restrict__ CwRe, _Float16* __restrict__ CwImN) {
  __shared__ float red[S4_Q][S4_N + 1];
  const int h = blockIdx.x, n = threadIdx.x;
  const float dt = expf(log_dt[h]);
  const float are = A_re[h * S4_N + n], aim = A_im[h * S4_N + n];
  const float dre = dt * are, dim = dt * aim;
  const float ew  = expf(dre);
  const float wre = ew * cosf(dim), wim = ew * sinf(dim);
  // (w-1)/A  with A = are + i*aim
  const float ar = wre - 1.0f, ai = wim;
  const float inv = 1.0f / (are * are + aim * aim);
  const float gr = (ar * are + ai * aim) * inv;
  const float gi = (ai * are - ar * aim) * inv;
  const float cr = C_re[h * S4_N + n], ci = C_im[h * S4_N + n];
  // Cf = 2 * C * (w-1)/A   (fold the 2*Re(.) factor here)
  const float Cfr = 2.0f * (cr * gr - ci * gi);
  const float Cfi = 2.0f * (cr * gi + ci * gr);

  float pr = 1.0f, pim = 0.0f;  // w^p, starting p=0
  const size_t base = (size_t)h * (S4_N * S4_Q);
  for (int p = 0; p < S4_Q; ++p) {
    // Wr[n][j] = w^{Q-1-j}  ->  j = 63-p
    WrRe[base + n * S4_Q + (63 - p)] = (_Float16)pr;
    WrIm[base + n * S4_Q + (63 - p)] = (_Float16)pim;
    // k[p] = sum_n Re(Cf * w^p)
    red[p][n] = Cfr * pr - Cfi * pim;
    // advance to w^{p+1}
    const float nr = pr * wre - pim * wim;
    const float ni = pr * wim + pim * wre;
    pr = nr; pim = ni;
    // Cw[r=p][n] = Cf * w^{p+1};  Im part pre-negated for pure WMMA accumulation
    CwRe [base + p * S4_N + n] = (_Float16)(Cfr * pr - Cfi * pim);
    CwImN[base + p * S4_N + n] = (_Float16)(-(Cfr * pim + Cfi * pr));
  }
  wQre[h * S4_N + n] = pr;   // w^64
  wQim[h * S4_N + n] = pim;
  __syncthreads();
  float s = 0.0f;
#pragma unroll 8
  for (int j = 0; j < S4_N; ++j) s += red[n][j];
  kq[h * S4_Q + n] = s;
}

// ---------------------------------------------------------------------------
// Kernel 2: chunked SSM conv + skip + GELU.  grid = B*H, block = 32 (one wave).
// ---------------------------------------------------------------------------
__global__ __launch_bounds__(32) void s4_chunk_scan(
    const float* __restrict__ u, const float* __restrict__ Dv,
    const float* __restrict__ kq,
    const float* __restrict__ wQre, const float* __restrict__ wQim,
    const _Float16* __restrict__ WrRe, const _Float16* __restrict__ WrIm,
    const _Float16* __restrict__ CwRe, const _Float16* __restrict__ CwImN,
    _Float16* __restrict__ G) {
  __shared__ __align__(16) float uf32[S4_L];     // U col-major f32: uf32[c*64+j] = u[l]
  __shared__ float    Sre[S4_L], Sim[S4_L];      // S[n,c] at c*64+n
  __shared__ _Float16 XpRe[S4_L], XpIm[S4_L];
  __shared__ float    kql[S4_Q];

  const int lane  = threadIdx.x;
  const int bh    = blockIdx.x;
  const int h     = bh & (S4_H - 1);
  const int laneM = lane & 15, kh = lane >> 4;

  const float* up = u + (size_t)bh * S4_L;

#if defined(__gfx1250__)
  // Async DMA: u (8 KB) -> LDS, no VGPR round-trip.  Tracked by ASYNCcnt.
  {
    const unsigned ldsbase = (unsigned)(size_t)(&uf32[0]);
#pragma unroll
    for (int i = 0; i < 16; ++i) {
      const unsigned loff = ldsbase + (unsigned)((i * 128 + lane * 4) * 4);
      const unsigned long long ga = (unsigned long long)(size_t)(up + i * 128 + lane * 4);
      asm volatile("global_load_async_to_lds_b128 %0, %1, off"
                   :: "v"(loff), "v"(ga) : "memory");
    }
    asm volatile("s_wait_asynccnt 0" ::: "memory");
  }
#else
#pragma unroll
  for (int i = 0; i < 16; ++i) {
    const float4 f = ((const float4*)up)[i * 32 + lane];
    *((float4*)(uf32 + (i * 32 + lane) * 4)) = f;
  }
#endif
  kql[lane]      = kq[h * S4_Q + lane];
  kql[lane + 32] = kq[h * S4_Q + lane + 32];
  __syncthreads();

  // U B-fragments (f32 LDS -> f16 fragments), reused by phases B and final.
  v16h ub[2][2];
#pragma unroll
  for (int kt = 0; kt < 2; ++kt)
#pragma unroll
    for (int ct = 0; ct < 2; ++ct)
      ub[kt][ct] = load_b_cm_f32(uf32, ct * 16 + laneM, kt * 32, kh, 64);

  // ---- Phase B: S_{re,im} = Wr_{re,im} @ U ----
  const _Float16* wr_re = WrRe + (size_t)h * 4096;
  const _Float16* wr_im = WrIm + (size_t)h * 4096;
#pragma unroll
  for (int part = 0; part < 2; ++part) {
    const _Float16* wsrc = part ? wr_im : wr_re;
    float* sdst = part ? Sim : Sre;
#pragma unroll
    for (int mt = 0; mt < 4; ++mt) {
      const v16h a0 = load_a_f16(wsrc + mt * 16 * 64, laneM, 0, kh, 64);
      const v16h a1 = load_a_f16(wsrc + mt * 16 * 64, laneM, 32, kh, 64);
#pragma unroll
      for (int ct = 0; ct < 2; ++ct) {
        v8f acc = {0.f, 0.f, 0.f, 0.f, 0.f, 0.f, 0.f, 0.f};
        acc = wmma_f16(a0, ub[0][ct], acc);
        acc = wmma_f16(a1, ub[1][ct], acc);
        const int c = ct * 16 + laneM;
        const int n = mt * 16 + kh * 8;
        *((float4*)(sdst + c * 64 + n))     = make_float4(acc[0], acc[1], acc[2], acc[3]);
        *((float4*)(sdst + c * 64 + n + 4)) = make_float4(acc[4], acc[5], acc[6], acc[7]);
      }
    }
  }
  __syncthreads();

  // ---- Phase C: 32-step chunk scan, 2 complex states per lane ----
  {
    const int n0 = lane * 2;
    float x0r = 0.f, x0i = 0.f, x1r = 0.f, x1i = 0.f;
    const float w0r = wQre[h * S4_N + n0],     w0i = wQim[h * S4_N + n0];
    const float w1r = wQre[h * S4_N + n0 + 1], w1i = wQim[h * S4_N + n0 + 1];
    for (int c = 0; c < S4_NC; ++c) {
      union { _Float16 h2[2]; unsigned u1; } pr, pi;
      pr.h2[0] = (_Float16)x0r; pr.h2[1] = (_Float16)x1r;
      pi.h2[0] = (_Float16)x0i; pi.h2[1] = (_Float16)x1i;
      *((unsigned*)(XpRe + c * 64 + n0)) = pr.u1;   // Xp[:,c] = X[c-1]
      *((unsigned*)(XpIm + c * 64 + n0)) = pi.u1;
      const float s0r = Sre[c * 64 + n0], s1r = Sre[c * 64 + n0 + 1];
      const float s0i = Sim[c * 64 + n0], s1i = Sim[c * 64 + n0 + 1];
      const float t0r = w0r * x0r - w0i * x0i + s0r;
      const float t0i = w0r * x0i + w0i * x0r + s0i;
      const float t1r = w1r * x1r - w1i * x1i + s1r;
      const float t1i = w1r * x1i + w1i * x1r + s1i;
      x0r = t0r; x0i = t0i; x1r = t1r; x1i = t1i;
    }
  }
  __syncthreads();

  // Xprev B-fragments.
  v16h xbr[2][2], xbi[2][2];
#pragma unroll
  for (int kt = 0; kt < 2; ++kt)
#pragma unroll
    for (int ct = 0; ct < 2; ++ct) {
      xbr[kt][ct] = load_b_cm(XpRe, ct * 16 + laneM, kt * 32, kh, 64);
      xbi[kt][ct] = load_b_cm(XpIm, ct * 16 + laneM, kt * 32, kh, 64);
    }

  // ---- Final: Y = CwRe@XpRe + CwImNeg@XpIm + T@U ; epilogue GELU ----
  const _Float16* cw_re = CwRe  + (size_t)h * 4096;
  const _Float16* cw_in = CwImN + (size_t)h * 4096;
  const float Dh = Dv[h];
  _Float16* gout = G + (size_t)bh * S4_L;

#pragma unroll
  for (int mt = 0; mt < 4; ++mt) {
    const v16h ar0 = load_a_f16(cw_re + mt * 16 * 64, laneM, 0, kh, 64);
    const v16h ar1 = load_a_f16(cw_re + mt * 16 * 64, laneM, 32, kh, 64);
    const v16h ai0 = load_a_f16(cw_in + mt * 16 * 64, laneM, 0, kh, 64);
    const v16h ai1 = load_a_f16(cw_in + mt * 16 * 64, laneM, 32, kh, 64);
    // Toeplitz A-fragments from k: T[r,j] = k[r-j] (j<=r)
    v16h at0, at1;
    {
      const int r = mt * 16 + laneM;
#pragma unroll
      for (int v = 0; v < 8; ++v) {
        const int kb = ((v >= 4) ? 16 : 0) + kh * 8 + (v & 3) * 2;
        const int d0 = r - kb, d1 = d0 - 1;
        at0[2 * v]     = (d0 >= 0) ? (_Float16)kql[d0] : (_Float16)0.f;
        at0[2 * v + 1] = (d1 >= 0) ? (_Float16)kql[d1] : (_Float16)0.f;
        const int e0 = d0 - 32, e1 = d1 - 32;
        at1[2 * v]     = (e0 >= 0) ? (_Float16)kql[e0] : (_Float16)0.f;
        at1[2 * v + 1] = (e1 >= 0) ? (_Float16)kql[e1] : (_Float16)0.f;
      }
    }
#pragma unroll
    for (int ct = 0; ct < 2; ++ct) {
      v8f acc = {0.f, 0.f, 0.f, 0.f, 0.f, 0.f, 0.f, 0.f};
      acc = wmma_f16(ar0, xbr[0][ct], acc);
      acc = wmma_f16(ar1, xbr[1][ct], acc);
      acc = wmma_f16(ai0, xbi[0][ct], acc);
      acc = wmma_f16(ai1, xbi[1][ct], acc);
      acc = wmma_f16(at0, ub[0][ct], acc);
      acc = wmma_f16(at1, ub[1][ct], acc);
      const int c = ct * 16 + laneM;
      const int rbase = mt * 16 + kh * 8;
      v8h gh;
#pragma unroll
      for (int v = 0; v < 8; ++v) {
        const int l = c * 64 + rbase + v;
        const float y = acc[v] + Dh * uf32[l];
        const float g = 0.5f * y * (1.0f + erff(y * 0.70710678f));
        gh[v] = (_Float16)g;
      }
      *((v8h*)(gout + c * 64 + rbase)) = gh;  // 16B store
    }
  }
}

// ---------------------------------------------------------------------------
// Kernel 3: z = W_out @ g + b, GLU.  grid = (L/64, H/64, B), block = 256.
// ---------------------------------------------------------------------------
__global__ __launch_bounds__(256) void s4_outproj(
    const _Float16* __restrict__ G, const float* __restrict__ W,
    const float* __restrict__ bias, float* __restrict__ out) {
  __shared__ _Float16 gt[64 * 32];   // g tile, col-major: (k,col) at col*32+k
  __shared__ float comb[64 * 64];    // sigmoid(h2) for GLU pairing

  const int lt = blockIdx.x, ot = blockIdx.y, b = blockIdx.z;
  const int l0 = lt * 64, ho = ot * 64;
  const int tid = threadIdx.x;
  const int wave = tid >> 5, lane = tid & 31;
  const int half = wave & 1, mt = wave >> 1;
  const int laneM = lane & 15, kh = lane >> 4;
  const int orow = half * 512 + ho + mt * 16;

  const v8f z8 = {0.f, 0.f, 0.f, 0.f, 0.f, 0.f, 0.f, 0.f};
  v8f acc[4] = {z8, z8, z8, z8};

  const _Float16* gb = G + (size_t)b * S4_H * S4_L;
  for (int ks = 0; ks < 16; ++ks) {
    __syncthreads();
    {
      const int col = tid & 63, kb = (tid >> 6) * 8;
      const _Float16* gsrc = gb + (size_t)(ks * 32 + kb) * S4_L + l0 + col;
#pragma unroll
      for (int kk = 0; kk < 8; ++kk)
        gt[col * 32 + kb + kk] = gsrc[(size_t)kk * S4_L];
      if (ks < 15)  // prefetch next 32-row g panel (speculative, L2-targeted)
        __builtin_prefetch(gsrc + (size_t)32 * S4_L, 0, 1);
    }
    __syncthreads();
    const v16h a = load_a_f32(W + (size_t)orow * S4_H + ks * 32, laneM, 0, kh, S4_H);
#pragma unroll
    for (int ct = 0; ct < 4; ++ct) {
      const v16h bf = load_b_cm(gt, ct * 16 + laneM, 0, kh, 32);
      acc[ct] = wmma_f16(a, bf, acc[ct]);
    }
  }
  __syncthreads();

  if (half == 1) {
#pragma unroll
    for (int ct = 0; ct < 4; ++ct)
#pragma unroll
      for (int v = 0; v < 8; ++v) {
        const int ol = mt * 16 + kh * 8 + v;
        const float z = acc[ct][v] + bias[512 + ho + ol];
        comb[ol * 64 + ct * 16 + laneM] = 1.0f / (1.0f + expf(-z));
      }
  }
  __syncthreads();
  if (half == 0) {
#pragma unroll
    for (int ct = 0; ct < 4; ++ct)
#pragma unroll
      for (int v = 0; v < 8; ++v) {
        const int ol = mt * 16 + kh * 8 + v;
        const float z = acc[ct][v] + bias[ho + ol];
        out[((size_t)b * S4_H + ho + ol) * S4_L + l0 + ct * 16 + laneM] =
            z * comb[ol * 64 + ct * 16 + laneM];
      }
  }
}

// ---------------------------------------------------------------------------
extern "C" void kernel_launch(void* const* d_in, const int* in_sizes, int n_in,
                              void* d_out, int out_size, void* d_ws, size_t ws_size,
                              hipStream_t stream) {
  const float* u      = (const float*)d_in[0];
  const float* log_dt = (const float*)d_in[1];
  const float* A_re   = (const float*)d_in[2];
  const float* A_im   = (const float*)d_in[3];
  const float* C_re   = (const float*)d_in[4];
  const float* C_im   = (const float*)d_in[5];
  const float* Dv     = (const float*)d_in[6];
  const float* W_out  = (const float*)d_in[7];
  const float* b_out  = (const float*)d_in[8];
  float* out = (float*)d_out;

  char* ws = (char*)d_ws;
  float*    kq    = (float*)(ws + 0);                         // 512*64 f32
  float*    wQre  = (float*)(ws + 131072);                    // 512*64 f32
  float*    wQim  = (float*)(ws + 262144);                    // 512*64 f32
  _Float16* WrRe  = (_Float16*)(ws + 393216);                 // 512*4096 f16
  _Float16* WrIm  = (_Float16*)(ws + 393216 + 1 * 4194304);
  _Float16* CwRe  = (_Float16*)(ws + 393216 + 2 * 4194304);
  _Float16* CwImN = (_Float16*)(ws + 393216 + 3 * 4194304);
  _Float16* G     = (_Float16*)(ws + 393216 + 4 * 4194304);   // 8*512*2048 f16

  s4_precompute<<<S4_H, S4_N, 0, stream>>>(log_dt, A_re, A_im, C_re, C_im,
                                           kq, wQre, wQim, WrRe, WrIm, CwRe, CwImN);
  s4_chunk_scan<<<S4_B * S4_H, 32, 0, stream>>>(u, Dv, kq, wQre, wQim,
                                                WrRe, WrIm, CwRe, CwImN, G);
  dim3 g3(S4_L / 64, S4_H / 64, S4_B);
  s4_outproj<<<g3, 256, 0, stream>>>(G, W_out, b_out, out);
}
